// BlockInvariantPointAttention_59115929862883
// MI455X (gfx1250) — matrix operand order repaired and checked by reading, not compiled, not run on hardware
//
#include <hip/hip_runtime.h>
#include <hip/hip_bf16.h>
#include <math.h>

#define BB   2
#define NN   2048
#define BQ   32
#define BKK  128
#define NBB  64
#define CS   384
#define CZ   128
#define CH   16
#define HH   12
#define PQK  4
#define PV   8
#define ROWS   (BB*NN)           // 4096
#define ZROWS  ((size_t)BB*NBB*BQ*BKK) // 524288
#define PROJC  1152
#define FEATC  960
#define VCS    48                // padded [v|v_pts] row stride (halves)

typedef _Float16 h16;
typedef __attribute__((ext_vector_type(16))) _Float16 v16h;
typedef __attribute__((ext_vector_type(8)))  _Float16 v8h;
typedef __attribute__((ext_vector_type(8)))  float    v8f;

// ---- WMMA helpers (ISA 16x16x32 f16 layouts) -------------------------------
__device__ __forceinline__ v16h ld_ab(const h16* lo, const h16* hi) {
  v8h a = *(const v8h*)lo;
  v8h b = *(const v8h*)hi;
  v16h r;
#pragma unroll
  for (int i = 0; i < 8; ++i) { r[i] = a[i]; r[i + 8] = b[i]; }
  return r;
}
// A tile: rows m0..m0+15, K window [k0,k0+32), row-major, leading dim ld
__device__ __forceinline__ v16h ld_atile(const h16* A, int ld, int m0, int k0, int lane) {
  int m = m0 + (lane & 15);
  int base = (lane >> 4) * 8;
  const h16* p = A + (size_t)m * ld + k0 + base;
  return ld_ab(p, p + 16);
}
// B tile: K rows k0..k0+31, cols n0..n0+15, row-major, leading dim ld
__device__ __forceinline__ v16h ld_btile(const h16* Bm, int ld, int k0, int n0, int lane) {
  const h16* p = Bm + (size_t)(k0 + (lane >> 4) * 8) * ld + n0 + (lane & 15);
  v16h r;
#pragma unroll
  for (int i = 0; i < 8; ++i) { r[i] = p[(size_t)i * ld]; r[i + 8] = p[(size_t)(16 + i) * ld]; }
  return r;
}
__device__ __forceinline__ v8f wmma16(v16h a, v16h b, v8f c) {
  return __builtin_amdgcn_wmma_f32_16x16x32_f16(false, a, false, b, (short)0, c, false, false);
}
__device__ __forceinline__ float red32_sum(float v) {
#pragma unroll
  for (int m = 1; m < 32; m <<= 1) v += __shfl_xor(v, m, 32);
  return v;
}
__device__ __forceinline__ float red8_sum(float v) {
#pragma unroll
  for (int m = 1; m < 8; m <<= 1) v += __shfl_xor(v, m, 32);
  return v;
}
__device__ __forceinline__ float red8_max(float v) {
#pragma unroll
  for (int m = 1; m < 8; m <<= 1) v = fmaxf(v, __shfl_xor(v, m, 32));
  return v;
}

// ---- K0: convert/pack weights to f16 ---------------------------------------
__global__ __launch_bounds__(256) void ipa_convert(
    const float* __restrict__ Wq, const float* __restrict__ Wk, const float* __restrict__ Wv,
    const float* __restrict__ Wqp, const float* __restrict__ Wkvp,
    const float* __restrict__ Wb, const float* __restrict__ Wdz, const float* __restrict__ Wout,
    h16* __restrict__ Wproj, h16* __restrict__ Wz, h16* __restrict__ Wouth)
{
  const int NP = 384 * 1152, NZ = 128 * 48, NO = 960 * 384;
  for (int i = blockIdx.x * 256 + threadIdx.x; i < NP + NZ + NO; i += gridDim.x * 256) {
    if (i < NP) {
      int r = i / 1152, c = i % 1152; float v;
      if      (c < 192) v = Wq [r * 192 + c];
      else if (c < 384) v = Wk [r * 192 + c - 192];
      else if (c < 576) v = Wv [r * 192 + c - 384];
      else if (c < 720) v = Wqp[r * 144 + c - 576];
      else              v = Wkvp[r * 432 + c - 720];
      Wproj[i] = (h16)v;
    } else if (i < NP + NZ) {
      int j = i - NP; int r = j / 48, c = j % 48; float v;
      if      (c < 12) v = Wb[r * 12 + c] * 0.5773502691896258f; // fold sqrt(1/3)
      else if (c < 16) v = 0.f;
      else             v = Wdz[r * 32 + c - 16];
      Wz[j] = (h16)v;
    } else {
      int j = i - NP - NZ;
      Wouth[j] = (h16)Wout[j];
    }
  }
}

// ---- K1: LayerNorm(s) -> f16 (wave per row) --------------------------------
__global__ __launch_bounds__(256) void ipa_ln_s(
    const float* __restrict__ s, const float* __restrict__ g, const float* __restrict__ bb,
    h16* __restrict__ sN)
{
  int lane = threadIdx.x & 31, wv = threadIdx.x >> 5;
  int row = blockIdx.x * 8 + wv;
  if (row >= ROWS) return;
  const float* sp = s + (size_t)row * CS;
  float x[12]; float sm = 0.f;
#pragma unroll
  for (int i = 0; i < 12; ++i) { x[i] = sp[lane + 32 * i]; sm += x[i]; }
  sm = red32_sum(sm);
  float mean = sm * (1.0f / 384.0f);
  float vs = 0.f;
#pragma unroll
  for (int i = 0; i < 12; ++i) { float d = x[i] - mean; vs += d * d; }
  vs = red32_sum(vs);
  float rstd = rsqrtf(vs * (1.0f / 384.0f) + 1e-5f);
#pragma unroll
  for (int i = 0; i < 12; ++i) {
    int c = lane + 32 * i;
    sN[(size_t)row * CS + c] = (h16)((x[i] - mean) * rstd * g[c] + bb[c]);
  }
}

// ---- K2: projection GEMM 4096x384 @ 384x1152 -> f32 ------------------------
__global__ __launch_bounds__(256) void ipa_proj_gemm(
    const h16* __restrict__ sN, const h16* __restrict__ Wproj, float* __restrict__ proj)
{
  int lane = threadIdx.x & 31, wv = threadIdx.x >> 5;
  int tid = blockIdx.x * 8 + wv;
  int mt = tid & 255, nt = tid >> 8;   // 256 x 72 tiles
  v8f acc = {};
#pragma unroll
  for (int ks = 0; ks < 12; ++ks) {
    v16h a = ld_atile(sN, CS, mt * 16, ks * 32, lane);
    v16h b = ld_btile(Wproj, PROJC, ks * 32, nt * 16, lane);
    acc = wmma16(a, b, acc);
  }
#pragma unroll
  for (int r = 0; r < 8; ++r) {
    int row = mt * 16 + r + ((lane >> 4) << 3);
    proj[(size_t)row * PROJC + nt * 16 + (lane & 15)] = acc[r];
  }
}

// ---- K3: rotations + pack attention operands (thread per row,head) ---------
__global__ __launch_bounds__(256) void ipa_prep(
    const float* __restrict__ proj, const float* __restrict__ rots,
    const float* __restrict__ trans, const float* __restrict__ hws,
    h16* __restrict__ qAT, h16* __restrict__ kB, h16* __restrict__ vcat)
{
  int tid = blockIdx.x * 256 + threadIdx.x;
  if (tid >= ROWS * HH) return;
  int row = tid / HH, h = tid % HH;
  const float* pr = proj + (size_t)row * PROJC;
  const float* R  = rots + (size_t)row * 9;
  const float* tr = trans + (size_t)row * 3;
  float hv = hws[h];
  float sp = (hv > 20.f) ? hv : log1pf(__expf(hv));
  float hw = sp * 0.13608276348795434f;    // sqrt(1/54)
  float cc = -0.5f * hw;
  const float s1 = 0.14433756729740643f;   // sqrt(1/48)

  float qr[PQK][3]; float qn = 0.f;
#pragma unroll
  for (int p = 0; p < PQK; ++p) {
    float v0 = pr[576 + (h * PQK + p) * 3 + 0];
    float v1 = pr[576 + (h * PQK + p) * 3 + 1];
    float v2 = pr[576 + (h * PQK + p) * 3 + 2];
#pragma unroll
    for (int i = 0; i < 3; ++i) {
      float x = R[i * 3 + 0] * v0 + R[i * 3 + 1] * v1 + R[i * 3 + 2] * v2 + tr[i];
      qr[p][i] = x; qn += x * x;
    }
  }
  float kr[PQK][3]; float kn = 0.f; float vp[PV][3];
#pragma unroll
  for (int p = 0; p < 12; ++p) {
    float v0 = pr[720 + (h * 12 + p) * 3 + 0];
    float v1 = pr[720 + (h * 12 + p) * 3 + 1];
    float v2 = pr[720 + (h * 12 + p) * 3 + 2];
#pragma unroll
    for (int i = 0; i < 3; ++i) {
      float x = R[i * 3 + 0] * v0 + R[i * 3 + 1] * v1 + R[i * 3 + 2] * v2 + tr[i];
      if (p < PQK) { kr[p][i] = x; kn += x * x; }
      else         vp[p - PQK][i] = x;
    }
  }
  int b = row >> 11; int n = row & 2047; int nb = n >> 5; int qq = n & 31;
  // qAT[b][nb][h][feat(32)][q(32)]
  h16* qa = qAT + ((size_t)((b * NBB + nb) * HH + h)) * 1024 + qq;
#pragma unroll
  for (int c = 0; c < 16; ++c) qa[c * 32] = (h16)(pr[h * 16 + c] * s1);
#pragma unroll
  for (int p = 0; p < PQK; ++p)
#pragma unroll
    for (int i = 0; i < 3; ++i) qa[(16 + p * 3 + i) * 32] = (h16)(qr[p][i] * hw);
  qa[28 * 32] = (h16)(cc * qn); qa[29 * 32] = (h16)cc;
  qa[30 * 32] = (h16)0.f;       qa[31 * 32] = (h16)0.f;
  // kB[row][h][32]
  h16* kb = kB + ((size_t)row * HH + h) * 32;
#pragma unroll
  for (int c = 0; c < 16; ++c) kb[c] = (h16)pr[192 + h * 16 + c];
#pragma unroll
  for (int p = 0; p < PQK; ++p)
#pragma unroll
    for (int i = 0; i < 3; ++i) kb[16 + p * 3 + i] = (h16)kr[p][i];
  kb[28] = (h16)1.f; kb[29] = (h16)kn; kb[30] = (h16)0.f; kb[31] = (h16)0.f;
  // vcat[row][h][48] = [v(16) | v_pts(24) | zero-pad(8)]  (16B-aligned rows)
  h16* vc = vcat + ((size_t)row * HH + h) * VCS;
#pragma unroll
  for (int c = 0; c < 16; ++c) vc[c] = (h16)pr[384 + h * 16 + c];
#pragma unroll
  for (int p = 0; p < PV; ++p)
#pragma unroll
    for (int i = 0; i < 3; ++i) vc[16 + p * 3 + i] = (h16)vp[p][i];
#pragma unroll
  for (int c = 40; c < 48; ++c) vc[c] = (h16)0.f;
}

// ---- K4: fused z pipeline: LN(z) -> [sqrt(1/3)Wb | Wdz] via WMMA ----------
__global__ __launch_bounds__(128) void ipa_zproj(
    const float* __restrict__ z, const float* __restrict__ gz, const float* __restrict__ bz,
    const h16* __restrict__ Wz, float* __restrict__ biasb, h16* __restrict__ pairz)
{
  __shared__ h16  zt[64 * 128];
  __shared__ float gsh[128], bsh[128];
  int t = threadIdx.x, lane = t & 31, wv = t >> 5;
  if (t < 128) { gsh[t] = gz[t]; bsh[t] = bz[t]; }
  __syncthreads();
  size_t row0 = (size_t)blockIdx.x * 64;
  for (int it = 0; it < 16; ++it) {
    int r = wv * 16 + it;
    const float* zp = z + (row0 + r) * CZ + lane * 4;
    if (it + 1 < 16)                 // hint the HBM stream ahead of the LN math
      __builtin_prefetch(zp + CZ, 0, 1);
    const float4 v4 = *(const float4*)zp;
    float sm = v4.x + v4.y + v4.z + v4.w;
    float sq = v4.x * v4.x + v4.y * v4.y + v4.z * v4.z + v4.w * v4.w;
    sm = red32_sum(sm); sq = red32_sum(sq);
    float mean = sm * (1.0f / 128.0f);
    float var  = sq * (1.0f / 128.0f) - mean * mean;
    float rstd = rsqrtf(var + 1e-5f);
    int c0 = lane * 4;
    zt[r * 128 + c0 + 0] = (h16)((v4.x - mean) * rstd * gsh[c0 + 0] + bsh[c0 + 0]);
    zt[r * 128 + c0 + 1] = (h16)((v4.y - mean) * rstd * gsh[c0 + 1] + bsh[c0 + 1]);
    zt[r * 128 + c0 + 2] = (h16)((v4.z - mean) * rstd * gsh[c0 + 2] + bsh[c0 + 2]);
    zt[r * 128 + c0 + 3] = (h16)((v4.w - mean) * rstd * gsh[c0 + 3] + bsh[c0 + 3]);
  }
  __syncthreads();
  for (int nt = 0; nt < 3; ++nt) {
    v8f acc = {};
#pragma unroll
    for (int ks = 0; ks < 4; ++ks) {
      v16h a = ld_atile(zt, 128, wv * 16, ks * 32, lane);
      v16h b = ld_btile(Wz, 48, ks * 32, nt * 16, lane);
      acc = wmma16(a, b, acc);
    }
#pragma unroll
    for (int r = 0; r < 8; ++r) {
      int m = r + ((lane >> 4) << 3);
      size_t zr = row0 + wv * 16 + m;
      int nidx = nt * 16 + (lane & 15);
      if (nt == 0) { if (nidx < HH) biasb[zr * HH + nidx] = acc[r]; }
      else pairz[zr * 32 + (nidx - 16)] = (h16)acc[r];
    }
  }
}

// ---- K5a: attention core per (b,nb): logits WMMA -> softmax -> o/o_pt ------
__global__ __launch_bounds__(256) void ipa_attn(
    const h16* __restrict__ qAT, const h16* __restrict__ kB, const h16* __restrict__ vcat,
    const float* __restrict__ biasb, const float* __restrict__ s_mask,
    const int* __restrict__ key_idx,
    h16* __restrict__ Pbuf, h16* __restrict__ feats, float* __restrict__ optR)
{
  __shared__ h16   lds_v[128 * VCS];
  __shared__ float lds_l[32 * 128];
  __shared__ h16   lds_p[32 * 128];
  __shared__ int   lds_kidx[128];
  int t = threadIdx.x, lane = t & 31, wv = t >> 5;
  int b = blockIdx.x >> 6, nb = blockIdx.x & 63;
  if (t < 128) lds_kidx[t] = key_idx[nb * BKK + t];
  __syncthreads();
  for (int h = 0; h < HH; ++h) {
    { // gather padded [v | v_pts] rows into LDS: 3 x b128 per thread
      int k = t >> 1; int cb = (t & 1) * 24;
      int g = b * NN + lds_kidx[k];
      const v8h* src = (const v8h*)(vcat + ((size_t)g * HH + h) * VCS + cb);
      v8h* dst = (v8h*)(lds_v + k * VCS + cb);
      dst[0] = src[0]; dst[1] = src[1]; dst[2] = src[2];
    }
    __syncthreads();
    // logits^T (k-major) : C = sqrt(1/3)*bias + INF*(mask-1)
#pragma unroll
    for (int rep = 0; rep < 2; ++rep) {
      int tt = wv + rep * 8;
      int kt = tt & 7, qt = tt >> 3;
      v8f c;
#pragma unroll
      for (int r = 0; r < 8; ++r) {
        int kk = kt * 16 + r + ((lane >> 4) << 3);
        int q  = qt * 16 + (lane & 15);
        size_t zr = ((size_t)((b * NBB + nb) * BQ + q)) * BKK + kk;
        float qm = s_mask[b * NN + nb * BQ + q];
        float km = s_mask[b * NN + lds_kidx[kk]];
        c[r] = biasb[zr * HH + h] + 100000.0f * (qm * km - 1.0f);
      }
      int kkA = kt * 16 + (lane & 15);
      int g = b * NN + lds_kidx[kkA];
      const h16* pk = kB + ((size_t)g * HH + h) * 32;
      int base = (lane >> 4) * 8;
      v16h a = ld_ab(pk + base, pk + 16 + base);
      const h16* pq = qAT + ((size_t)((b * NBB + nb) * HH + h)) * 1024 + base * 32 + qt * 16 + (lane & 15);
      v16h bm;
#pragma unroll
      for (int i = 0; i < 8; ++i) { bm[i] = pq[i * 32]; bm[i + 8] = pq[(16 + i) * 32]; }
      v8f d = wmma16(a, bm, c);
#pragma unroll
      for (int r = 0; r < 8; ++r) {
        int kk = kt * 16 + r + ((lane >> 4) << 3);
        int q  = qt * 16 + (lane & 15);
        lds_l[q * 128 + kk] = d[r];   // free transpose via D scatter
      }
    }
    __syncthreads();
    { // softmax over k per (q); store normalized P in f16
      int q = t >> 3, k0 = (t & 7) * 16;
      float x[16]; float m = -3.0e38f;
#pragma unroll
      for (int j = 0; j < 16; ++j) { x[j] = lds_l[q * 128 + k0 + j]; m = fmaxf(m, x[j]); }
      m = red8_max(m);
      float ssum = 0.f;
#pragma unroll
      for (int j = 0; j < 16; ++j) { x[j] = __expf(x[j] - m); ssum += x[j]; }
      ssum = red8_sum(ssum);
      float inv = 1.0f / ssum;
      v8h p0, p1;
#pragma unroll
      for (int j = 0; j < 8; ++j) { p0[j] = (h16)(x[j] * inv); p1[j] = (h16)(x[j + 8] * inv); }
      *(v8h*)&lds_p[q * 128 + k0]     = p0;
      *(v8h*)&lds_p[q * 128 + k0 + 8] = p1;
      h16* gp = Pbuf + ((size_t)((b * NBB + nb) * HH + h)) * 4096 + q * 128 + k0;
      *(v8h*)gp = p0; *(v8h*)(gp + 8) = p1;
    }
    __syncthreads();
    // o (cols 0..15) and o_pt (cols 16..39) : P @ [v | v_pts]
    if (wv < 6) {
      int qt = wv & 1, ct = wv >> 1;
      v8f acc = {};
#pragma unroll
      for (int ks = 0; ks < 4; ++ks) {
        v16h a  = ld_atile(lds_p, 128, qt * 16, ks * 32, lane);
        v16h bm = ld_btile(lds_v, VCS, ks * 32, ct * 16, lane);
        acc = wmma16(a, bm, acc);
      }
#pragma unroll
      for (int r = 0; r < 8; ++r) {
        int q = qt * 16 + r + ((lane >> 4) << 3);
        int orow = (b * NBB + nb) * BQ + q;
        int c = ct * 16 + (lane & 15);
        if (ct == 0) feats[(size_t)orow * FEATC + h * CH + c] = (h16)acc[r];
        else { int cx = c - 16; if (cx < 24) optR[((size_t)orow * HH + h) * 24 + cx] = acc[r]; }
      }
    }
    __syncthreads();
  }
}

// ---- K5b: o_pair = P(12x128) @ pair_z(128x32) per query (wave per q) -------
__global__ __launch_bounds__(256) void ipa_opair(
    const h16* __restrict__ Pbuf, const h16* __restrict__ pairz, h16* __restrict__ feats)
{
  int t = threadIdx.x, lane = t & 31, wv = t >> 5;
  int oq = blockIdx.x * 8 + wv;            // 0..4095
  int b = oq >> 11; int r2 = oq & 2047; int nb = r2 >> 5; int qq = r2 & 31;
  int hidx = lane & 15;
  int base = (lane >> 4) * 8;
  const h16* pz = pairz + (size_t)oq * 128 * 32;
#pragma unroll
  for (int ct = 0; ct < 2; ++ct) {
    v8f acc = {};
#pragma unroll
    for (int ks = 0; ks < 4; ++ks) {
      v16h a;
      if (hidx < HH) {
        const h16* pa = Pbuf + (((size_t)((b * NBB + nb) * HH + hidx)) * BQ + qq) * BKK
                        + ks * 32 + base;
        a = ld_ab(pa, pa + 16);
      } else {
#pragma unroll
        for (int i = 0; i < 16; ++i) a[i] = (h16)0.f;
      }
      v16h bm = ld_btile(pz, 32, ks * 32, ct * 16, lane);
      acc = wmma16(a, bm, acc);
    }
#pragma unroll
    for (int r = 0; r < 8; ++r) {
      int hh = r + ((lane >> 4) << 3);
      int c = ct * 16 + (lane & 15);
      if (hh < HH) feats[(size_t)oq * FEATC + 576 + hh * 32 + c] = (h16)acc[r];
    }
  }
}

// ---- K6: inverse rotation of o_pt + norms -> feats -------------------------
__global__ __launch_bounds__(256) void ipa_post(
    const float* __restrict__ optR, const float* __restrict__ rots,
    const float* __restrict__ trans, h16* __restrict__ feats)
{
  int tid = blockIdx.x * 256 + threadIdx.x;
  if (tid >= ROWS * HH) return;
  int row = tid / HH, h = tid % HH;
  const float* R  = rots + (size_t)row * 9;
  const float* tr = trans + (size_t)row * 3;
  const float* op = optR + ((size_t)row * HH + h) * 24;
  h16* ff = feats + (size_t)row * FEATC;
#pragma unroll
  for (int p = 0; p < PV; ++p) {
    float x0 = op[p * 3 + 0] - tr[0];
    float x1 = op[p * 3 + 1] - tr[1];
    float x2 = op[p * 3 + 2] - tr[2];
    float o0 = R[0] * x0 + R[3] * x1 + R[6] * x2;  // R^T x
    float o1 = R[1] * x0 + R[4] * x1 + R[7] * x2;
    float o2 = R[2] * x0 + R[5] * x1 + R[8] * x2;
    ff[192 + h * 24 + p * 3 + 0] = (h16)o0;
    ff[192 + h * 24 + p * 3 + 1] = (h16)o1;
    ff[192 + h * 24 + p * 3 + 2] = (h16)o2;
    ff[480 + h * PV + p] = (h16)sqrtf(o0 * o0 + o1 * o1 + o2 * o2 + 1e-8f);
  }
}

// ---- K7: out = feats(4096x960) @ Wout(960x384) -----------------------------
__global__ __launch_bounds__(256) void ipa_out_gemm(
    const h16* __restrict__ feats, const h16* __restrict__ Wouth, float* __restrict__ out)
{
  int lane = threadIdx.x & 31, wv = threadIdx.x >> 5;
  int tid = blockIdx.x * 8 + wv;
  int mt = tid & 255, nt = tid >> 8;   // 256 x 24 tiles
  v8f acc = {};
#pragma unroll
  for (int ks = 0; ks < 30; ++ks) {
    v16h a = ld_atile(feats, FEATC, mt * 16, ks * 32, lane);
    v16h b = ld_btile(Wouth, CS, ks * 32, nt * 16, lane);
    acc = wmma16(a, b, acc);
  }
#pragma unroll
  for (int r = 0; r < 8; ++r) {
    int row = mt * 16 + r + ((lane >> 4) << 3);
    out[(size_t)row * CS + nt * 16 + (lane & 15)] = acc[r];
  }
}

// ---- host ------------------------------------------------------------------
extern "C" void kernel_launch(void* const* d_in, const int* in_sizes, int n_in,
                              void* d_out, int out_size, void* d_ws, size_t ws_size,
                              hipStream_t stream) {
  const float* s       = (const float*)d_in[0];
  const float* z       = (const float*)d_in[1];
  const float* trans   = (const float*)d_in[2];
  const float* rots    = (const float*)d_in[3];
  const float* s_mask  = (const float*)d_in[4];
  const int*   key_idx = (const int*)d_in[5];
  const float* ln_s_g  = (const float*)d_in[6];
  const float* ln_s_b  = (const float*)d_in[7];
  const float* ln_z_g  = (const float*)d_in[8];
  const float* ln_z_b  = (const float*)d_in[9];
  const float* Wq   = (const float*)d_in[10];
  const float* Wk   = (const float*)d_in[11];
  const float* Wv   = (const float*)d_in[12];
  const float* Wqp  = (const float*)d_in[13];
  const float* Wkvp = (const float*)d_in[14];
  const float* Wb   = (const float*)d_in[15];
  const float* Wdz  = (const float*)d_in[16];
  const float* hws  = (const float*)d_in[17];
  const float* Wout = (const float*)d_in[18];
  float* out = (float*)d_out;

  char* w = (char*)d_ws;
  size_t off = 0;
  auto alloc = [&](size_t bytes) -> void* {
    void* p = w + off;
    off = (off + bytes + 255) & ~(size_t)255;
    return p;
  };
  h16*   sN    = (h16*)  alloc((size_t)ROWS * CS * 2);
  h16*   Wproj = (h16*)  alloc((size_t)384 * 1152 * 2);
  h16*   Wz    = (h16*)  alloc((size_t)128 * 48 * 2);
  h16*   Wouth = (h16*)  alloc((size_t)960 * 384 * 2);
  float* proj  = (float*)alloc((size_t)ROWS * PROJC * 4);
  h16*   qAT   = (h16*)  alloc((size_t)BB * NBB * HH * 32 * 32 * 2);
  h16*   kB    = (h16*)  alloc((size_t)ROWS * HH * 32 * 2);
  h16*   vcat  = (h16*)  alloc((size_t)ROWS * HH * VCS * 2);
  float* biasb = (float*)alloc(ZROWS * HH * 4);
  h16*   pairz = (h16*)  alloc(ZROWS * 32 * 2);
  h16*   Pbuf  = (h16*)  alloc((size_t)BB * NBB * HH * BQ * BKK * 2);
  float* optR  = (float*)alloc((size_t)ROWS * HH * 24 * 4);
  h16*   feats = (h16*)  alloc((size_t)ROWS * FEATC * 2);
  (void)ws_size; (void)in_sizes; (void)n_in; (void)out_size;

  ipa_convert  <<<1024, 256, 0, stream>>>(Wq, Wk, Wv, Wqp, Wkvp, Wb, Wdz, Wout,
                                          Wproj, Wz, Wouth);
  ipa_ln_s     <<<ROWS / 8, 256, 0, stream>>>(s, ln_s_g, ln_s_b, sN);
  ipa_proj_gemm<<<(256 * 72) / 8, 256, 0, stream>>>(sN, Wproj, proj);
  ipa_prep     <<<(ROWS * HH) / 256, 256, 0, stream>>>(proj, rots, trans, hws, qAT, kB, vcat);
  ipa_zproj    <<<(int)(ZROWS / 64), 128, 0, stream>>>(z, ln_z_g, ln_z_b, Wz, biasb, pairz);
  ipa_attn     <<<BB * NBB, 256, 0, stream>>>(qAT, kB, vcat, biasb, s_mask, key_idx,
                                              Pbuf, feats, optR);
  ipa_opair    <<<(BB * NN) / 8, 256, 0, stream>>>(Pbuf, pairz, feats);
  ipa_post     <<<(ROWS * HH) / 256, 256, 0, stream>>>(optR, rots, trans, feats);
  ipa_out_gemm <<<(256 * 24) / 8, 256, 0, stream>>>(feats, Wouth, out);
}
